// P6Experts_773094113556
// MI455X (gfx1250) — matrix operand order repaired and checked by reading, not compiled, not run on hardware
//
#include <hip/hip_runtime.h>

#define T_TOK 8192
#define HDIM  2048
#define FDIM  5632

typedef __bf16 v8bf  __attribute__((ext_vector_type(8)));
typedef __bf16 v16bf __attribute__((ext_vector_type(16)));
typedef float  v8f   __attribute__((ext_vector_type(8)));
typedef float  v16f  __attribute__((ext_vector_type(16)));

#define LDR  40            // padded LDS row length (80B stride -> conflict-free)
#define TILE (128 * LDR)   // one 128x32 bf16 tile (padded), in elements

// ---- staging helpers -------------------------------------------------------

struct F16 { float4 x, y, z, w; };   // 16 raw fp32 in regs (prefetch)
struct U8b { uint4 a, b; };          // 16 bf16 in regs (prefetch)

__device__ __forceinline__ F16 ldg16(const float* __restrict__ src) {
    const float4* s4 = (const float4*)src;
    F16 r; r.x = s4[0]; r.y = s4[1]; r.z = s4[2]; r.w = s4[3]; return r;
}

__device__ __forceinline__ U8b ldg16bf(const unsigned short* __restrict__ src) {
    const uint4* s4 = (const uint4*)src;
    U8b r; r.a = s4[0]; r.b = s4[1]; return r;
}

// convert 16 fp32 -> 16 bf16 (v_cvt_pk_bf16_f32) and store 32B to LDS
__device__ __forceinline__ void st_cvt16(unsigned short* dst, const F16& f) {
    v16f v;
    v[0]  = f.x.x; v[1]  = f.x.y; v[2]  = f.x.z; v[3]  = f.x.w;
    v[4]  = f.y.x; v[5]  = f.y.y; v[6]  = f.y.z; v[7]  = f.y.w;
    v[8]  = f.z.x; v[9]  = f.z.y; v[10] = f.z.z; v[11] = f.z.w;
    v[12] = f.w.x; v[13] = f.w.y; v[14] = f.w.z; v[15] = f.w.w;
    v16bf b = __builtin_convertvector(v, v16bf);
    v8bf lo = __builtin_shufflevector(b, b, 0,1,2,3,4,5,6,7);
    v8bf hi = __builtin_shufflevector(b, b, 8,9,10,11,12,13,14,15);
    *(v8bf*)(dst)     = lo;
    *(v8bf*)(dst + 8) = hi;
}

__device__ __forceinline__ void st16bf(unsigned short* dst, const U8b& u) {
    ((uint4*)dst)[0] = u.a;
    ((uint4*)dst)[1] = u.b;
}

// ---- fragment loads (ISA §7.12.2 layouts) ---------------------------------

// A-fragment (16x32 bf16): lanes%16 = M rows; per-lane K = [kb,kb+8) ++ [kb+16,kb+24), kb=(lane/16)*8
__device__ __forceinline__ v16bf load_a_frag(const unsigned short* tile, int m0, int lane) {
    int row = m0 + (lane & 15);
    int kb  = (lane >> 4) * 8;
    const unsigned short* p = tile + row * LDR + kb;
    v8bf lo = *(const v8bf*)(p);
    v8bf hi = *(const v8bf*)(p + 16);
    return __builtin_shufflevector(lo, hi, 0,1,2,3,4,5,6,7,8,9,10,11,12,13,14,15);
}

// B-fragment (32x16 bf16): lanes%16 = N cols; per-lane K = [kb,kb+8) ++ [kb+16,kb+24), kb=(lane/16)*8
__device__ __forceinline__ v16bf load_b_frag(const unsigned short* tile, int n0, int lane) {
    int n  = n0 + (lane & 15);
    int kb = (lane >> 4) * 8;
    const unsigned short* p = tile + n * LDR + kb;
    v8bf lo = *(const v8bf*)(p);
    v8bf hi = *(const v8bf*)(p + 16);
    return __builtin_shufflevector(lo, hi, 0,1,2,3,4,5,6,7,8,9,10,11,12,13,14,15);
}

__device__ __forceinline__ unsigned short f2bf_rne(float f) {
    unsigned int u = __float_as_uint(f);
    u += 0x7FFFu + ((u >> 16) & 1u);
    return (unsigned short)(u >> 16);
}

// ---------------------------------------------------------------------------
// Kernel 1: fused  silu(X W1^T) * (X W2^T)  -> act[T,F] (bf16)
// block = 256 thr (8 waves), tile 128(M) x 128(F), wave = 64x32, K-step 32
// double-buffered LDS, one barrier per K-step, register global prefetch
// ---------------------------------------------------------------------------
__global__ void __launch_bounds__(256)
fc1_swiglu_kernel(const int* __restrict__ eptr,
                  const float* __restrict__ X,
                  const float* __restrict__ fc1_1,
                  const float* __restrict__ fc1_2,
                  unsigned short* __restrict__ act)
{
    __shared__ __align__(16) unsigned short As [2 * TILE];
    __shared__ __align__(16) unsigned short B1s[2 * TILE];
    __shared__ __align__(16) unsigned short B2s[2 * TILE];

    const int e = *eptr;
    const float* __restrict__ W1 = fc1_1 + (size_t)e * FDIM * HDIM;
    const float* __restrict__ W2 = fc1_2 + (size_t)e * FDIM * HDIM;

    const int fBase   = blockIdx.x * 128;
    const int rowBase = blockIdx.y * 128;
    const int tid  = threadIdx.x;
    const int lane = tid & 31;
    const int wave = tid >> 5;
    const int wm = wave >> 2;   // 0..1 -> 64 rows
    const int wn = wave & 3;    // 0..3 -> 32 cols

    const int sr = tid >> 1;         // staging row 0..127
    const int sh = (tid & 1) * 16;   // staging K offset

    const float* pXs  = X  + (size_t)(rowBase + sr) * HDIM + sh;
    const float* pW1s = W1 + (size_t)(fBase   + sr) * HDIM + sh;
    const float* pW2s = W2 + (size_t)(fBase   + sr) * HDIM + sh;
    unsigned short* lAs  = &As [sr * LDR + sh];
    unsigned short* lB1s = &B1s[sr * LDR + sh];
    unsigned short* lB2s = &B2s[sr * LDR + sh];

    v8f accG[4][2];
    v8f accU[4][2];
#pragma unroll
    for (int i = 0; i < 4; ++i)
#pragma unroll
        for (int j = 0; j < 2; ++j) { accG[i][j] = (v8f){}; accU[i][j] = (v8f){}; }

    // prologue: stage slab 0 into buffer 0
    {
        F16 pa  = ldg16(pXs);
        F16 pb1 = ldg16(pW1s);
        F16 pb2 = ldg16(pW2s);
        st_cvt16(lAs,  pa);
        st_cvt16(lB1s, pb1);
        st_cvt16(lB2s, pb2);
    }
    __syncthreads();

    int p = 0;
    for (int k0 = 0; k0 < HDIM; k0 += 32) {
        const bool more = (k0 + 32) < HDIM;

        F16 pa, pb1, pb2;
        if (more) {                       // global prefetch of slab k+1
            pa  = ldg16(pXs  + k0 + 32);
            pb1 = ldg16(pW1s + k0 + 32);
            pb2 = ldg16(pW2s + k0 + 32);
        }

        const unsigned short* cA  = As  + p * TILE;
        const unsigned short* cB1 = B1s + p * TILE;
        const unsigned short* cB2 = B2s + p * TILE;

        v16bf a[4], b1[2], b2[2];
#pragma unroll
        for (int mi = 0; mi < 4; ++mi)
            a[mi] = load_a_frag(cA, wm * 64 + mi * 16, lane);
#pragma unroll
        for (int ni = 0; ni < 2; ++ni) {
            b1[ni] = load_b_frag(cB1, wn * 32 + ni * 16, lane);
            b2[ni] = load_b_frag(cB2, wn * 32 + ni * 16, lane);
        }

#pragma unroll
        for (int mi = 0; mi < 4; ++mi)
#pragma unroll
            for (int ni = 0; ni < 2; ++ni) {
                accG[mi][ni] = __builtin_amdgcn_wmma_f32_16x16x32_bf16(
                    false, a[mi], false, b1[ni], (short)0, accG[mi][ni], false, false);
                accU[mi][ni] = __builtin_amdgcn_wmma_f32_16x16x32_bf16(
                    false, a[mi], false, b2[ni], (short)0, accU[mi][ni], false, false);
            }

        if (more) {                       // stage slab k+1 while WMMAs drain
            const int q = p ^ 1;
            st_cvt16(lAs  + q * TILE, pa);
            st_cvt16(lB1s + q * TILE, pb1);
            st_cvt16(lB2s + q * TILE, pb2);
        }
        __syncthreads();
        p ^= 1;
    }

    // epilogue: silu(g)*u -> bf16 act   (fast rcp: error << bf16 noise floor)
    const int l  = lane & 15;
    const int hs = lane >> 4;
#pragma unroll
    for (int mi = 0; mi < 4; ++mi)
#pragma unroll
        for (int ni = 0; ni < 2; ++ni)
#pragma unroll
            for (int r = 0; r < 8; ++r) {
                float g = accG[mi][ni][r];
                float u = accU[mi][ni][r];
                float s = g * __builtin_amdgcn_rcpf(1.0f + __expf(-g));
                int row = rowBase + wm * 64 + mi * 16 + r + 8 * hs;
                int col = fBase   + wn * 32 + ni * 16 + l;
                act[(size_t)row * FDIM + col] = f2bf_rne(s * u);
            }
}

// ---------------------------------------------------------------------------
// Kernel 2: out[T,H] (f32) = act[T,F](bf16) x W3[H,F]^T
// ---------------------------------------------------------------------------
__global__ void __launch_bounds__(256)
fc2_kernel(const int* __restrict__ eptr,
           const unsigned short* __restrict__ act,
           const float* __restrict__ fc2w,
           float* __restrict__ out)
{
    __shared__ __align__(16) unsigned short As[2 * TILE];
    __shared__ __align__(16) unsigned short Bs[2 * TILE];

    const int e = *eptr;
    const float* __restrict__ W3 = fc2w + (size_t)e * HDIM * FDIM;

    const int nBase   = blockIdx.x * 128;  // H
    const int rowBase = blockIdx.y * 128;  // T
    const int tid  = threadIdx.x;
    const int lane = tid & 31;
    const int wave = tid >> 5;
    const int wm = wave >> 2;
    const int wn = wave & 3;

    const int sr = tid >> 1;
    const int sh = (tid & 1) * 16;

    const unsigned short* pAs = act + (size_t)(rowBase + sr) * FDIM + sh;
    const float*          pBs = W3  + (size_t)(nBase   + sr) * FDIM + sh;
    unsigned short* lAs = &As[sr * LDR + sh];
    unsigned short* lBs = &Bs[sr * LDR + sh];

    v8f acc[4][2];
#pragma unroll
    for (int i = 0; i < 4; ++i)
#pragma unroll
        for (int j = 0; j < 2; ++j) acc[i][j] = (v8f){};

    {
        U8b pa = ldg16bf(pAs);
        F16 pb = ldg16(pBs);
        st16bf  (lAs, pa);
        st_cvt16(lBs, pb);
    }
    __syncthreads();

    int p = 0;
    for (int k0 = 0; k0 < FDIM; k0 += 32) {
        const bool more = (k0 + 32) < FDIM;

        U8b pa; F16 pb;
        if (more) {
            pa = ldg16bf(pAs + k0 + 32);
            pb = ldg16  (pBs + k0 + 32);
        }

        const unsigned short* cA = As + p * TILE;
        const unsigned short* cB = Bs + p * TILE;

        v16bf a[4], b[2];
#pragma unroll
        for (int mi = 0; mi < 4; ++mi)
            a[mi] = load_a_frag(cA, wm * 64 + mi * 16, lane);
#pragma unroll
        for (int ni = 0; ni < 2; ++ni)
            b[ni] = load_b_frag(cB, wn * 32 + ni * 16, lane);

#pragma unroll
        for (int mi = 0; mi < 4; ++mi)
#pragma unroll
            for (int ni = 0; ni < 2; ++ni)
                acc[mi][ni] = __builtin_amdgcn_wmma_f32_16x16x32_bf16(
                    false, a[mi], false, b[ni], (short)0, acc[mi][ni], false, false);

        if (more) {
            const int q = p ^ 1;
            st16bf  (lAs + q * TILE, pa);
            st_cvt16(lBs + q * TILE, pb);
        }
        __syncthreads();
        p ^= 1;
    }

    const int l  = lane & 15;
    const int hs = lane >> 4;
#pragma unroll
    for (int mi = 0; mi < 4; ++mi)
#pragma unroll
        for (int ni = 0; ni < 2; ++ni)
#pragma unroll
            for (int r = 0; r < 8; ++r) {
                int row = rowBase + wm * 64 + mi * 16 + r + 8 * hs;
                int col = nBase   + wn * 32 + ni * 16 + l;
                out[(size_t)row * HDIM + col] = acc[mi][ni][r];
            }
}

extern "C" void kernel_launch(void* const* d_in, const int* in_sizes, int n_in,
                              void* d_out, int out_size, void* d_ws, size_t ws_size,
                              hipStream_t stream) {
    (void)in_sizes; (void)n_in; (void)out_size; (void)ws_size;
    const int*   eidx = (const int*)  d_in[0];
    const float* X    = (const float*)d_in[1];
    const float* w11  = (const float*)d_in[2];
    const float* w12  = (const float*)d_in[3];
    const float* w2   = (const float*)d_in[4];
    float* out = (float*)d_out;
    unsigned short* act = (unsigned short*)d_ws;  // T*F bf16 = 92.3 MB scratch

    dim3 blk(256);
    dim3 g1(FDIM / 128, T_TOK / 128);
    fc1_swiglu_kernel<<<g1, blk, 0, stream>>>(eidx, X, w11, w12, act);
    dim3 g2(HDIM / 128, T_TOK / 128);
    fc2_kernel<<<g2, blk, 0, stream>>>(eidx, act, w2, out);
}
